// SimpleRNN_42726334660622
// MI455X (gfx1250) — matrix operand order repaired and checked by reading, not compiled, not run on hardware
//
#include <hip/hip_runtime.h>
#include <hip/hip_bf16.h>

// ---------------------------------------------------------------------------
// SimpleRNN for MI455X (gfx1250, wave32, WMMA).
//   Phase 1: xproj[b*T+t, :] = embed[x[b,t]] @ Wx + b_fc   (big parallel GEMM)
//   Phase 2: 4 workgroups, 16 batch rows each, scan t=0..T-1:
//            h = tanh(xproj_t + h @ Wh); Wh lives in VGPRs, h in LDS ping-pong.
//            xproj tile for t+1 is register-prefetched; WMMA chain split in 2;
//            tanh via native v_exp_f32/v_rcp_f32 (co-executes with WMMA).
//   Phase 3: out = h_last @ W_out + b_out
// All matmuls use v_wmma_f32_16x16x32_bf16 (bf16 A/B, f32 accumulate).
// ---------------------------------------------------------------------------

typedef __attribute__((ext_vector_type(16))) __bf16 v16bf;
typedef __attribute__((ext_vector_type(8)))  float  v8f;

constexpr int kB   = 64;
constexpr int kT   = 2048;
constexpr int kH   = 256;
constexpr int kOUT = 256;

__device__ __forceinline__ v8f wmma_bf16(v16bf a, v16bf b, v8f c) {
  // 8 args: (neg_a, A, neg_b, B, c_mod, C, reuse_a, reuse_b)
  return __builtin_amdgcn_wmma_f32_16x16x32_bf16(false, a, false, b, (short)0, c,
                                                 false, false);
}

// tanh(x) = sign(x) * (1 - e) / (1 + e),  e = exp2(-2*log2(e)*|x|).
// Uses v_exp_f32 / v_rcp_f32 natives; TRANS ops co-execute with WMMA.
__device__ __forceinline__ float fast_tanh(float x) {
  float ax = __builtin_fabsf(x);
  float e  = __builtin_amdgcn_exp2f(-2.8853900817779268f * ax);
  float r  = (1.0f - e) * __builtin_amdgcn_rcpf(1.0f + e);
  return __builtin_copysignf(r, x);
}

union AChunk { v16bf v; uint4 u[2]; };

// A 16x32 bf16 chunk (contiguous K, matches B): lane holds row M=lr; lanes 0-15 get
// K = k0..k0+15, lanes 16-31 get K = k0+16..k0+31.
// arow points at this lane's row in a row-major bf16 buffer (row stride kH).
__device__ __forceinline__ v16bf load_a_chunk(const __bf16* arow, int k0, int half) {
  AChunk a;
  a.u[0] = *(const uint4*)(arow + k0 + half * 16);
  a.u[1] = *(const uint4*)(arow + k0 + half * 16 + 8);
  return a.v;
}

// B 32x16 bf16 chunk: lane holds one column n; lanes 0-15 K=k0..k0+15,
// lanes 16-31 K=k0+16..k0+31 (caller folds 'half' into k0). Converted f32->bf16.
__device__ __forceinline__ v16bf load_b_chunk_f32(const float* __restrict__ W,
                                                  int ld, int k0, int n) {
  v16bf v;
  #pragma unroll
  for (int i = 0; i < 16; ++i) v[i] = (__bf16)W[(k0 + i) * ld + n];
  return v;
}

// ---------------------------------------------------------------------------
// Phase 1: gather + GEMM.  Grid: (B*T/128) workgroups of 512 threads (16 waves).
// Each WG: 128 M-rows x all 256 N-cols; wave w owns N-tile w; 8 M-subtiles each.
// ---------------------------------------------------------------------------
__global__ void __launch_bounds__(512)
rnn_phase1_xproj(const int* __restrict__ x, const float* __restrict__ embed,
                 const float* __restrict__ W_fc, const float* __restrict__ b_fc,
                 float* __restrict__ xproj) {
  extern __shared__ char smem[];
  __bf16* Abuf = (__bf16*)smem;                    // 128 x 256 bf16 = 64 KB
  int*    toks = (int*)(smem + 128 * kH * 2);      // 128 token ids

  const int tid  = threadIdx.x;
  const int wave = tid >> 5, lane = tid & 31;
  const int lr   = lane & 15, half = lane >> 4;
  const int n0   = wave * 16;
  const long rbase = (long)blockIdx.x * 128;       // flattened (b*T+t) row base

  // Wx = W_fc[0:256, :] as 8 bf16 B-chunks held in VGPRs for the whole block.
  v16bf Breg[8];
  #pragma unroll
  for (int c = 0; c < 8; ++c)
    Breg[c] = load_b_chunk_f32(W_fc, kH, c * 32 + half * 16, n0 + lr);
  const float bias = b_fc[n0 + lr];

  if (tid < 128) toks[tid] = x[rbase + tid];
  __syncthreads();

  // Gather 128 embedding rows -> bf16 LDS (coalesced along H).
  for (int idx = tid; idx < 128 * kH; idx += 512) {
    int row = idx >> 8, col = idx & (kH - 1);
    Abuf[idx] = (__bf16)embed[(long)toks[row] * kH + col];
  }
  __syncthreads();

  #pragma unroll 1
  for (int mt = 0; mt < 8; ++mt) {
    const __bf16* arow = Abuf + (mt * 16 + lr) * kH;
    v8f acc = {bias, bias, bias, bias, bias, bias, bias, bias};
    #pragma unroll
    for (int c = 0; c < 8; ++c)
      acc = wmma_bf16(load_a_chunk(arow, c * 32, half), Breg[c], acc);
    const long R0 = rbase + mt * 16;
    #pragma unroll
    for (int r = 0; r < 8; ++r)          // C layout: (M = r + 8*half, N = lr)
      xproj[(R0 + r + half * 8) * (long)kH + n0 + lr] = acc[r];
  }
}

// ---------------------------------------------------------------------------
// Phase 2: recurrent scan.  Grid: 4 workgroups of 512 threads; WG g owns batch
// rows [16g, 16g+16).  Wh (256x256) lives in VGPRs (8 chunks/wave); h ping-pongs
// between two 16x256 bf16 LDS buffers; one barrier per timestep.
// ---------------------------------------------------------------------------
__global__ void __launch_bounds__(512)
rnn_phase2_scan(const float* __restrict__ W_fc, const float* __restrict__ xproj,
                float* __restrict__ hs, float* __restrict__ hlast) {
  extern __shared__ char smem[];
  __bf16* hbuf0 = (__bf16*)smem;                   // 16 x 256 bf16
  __bf16* hbuf1 = hbuf0 + 16 * kH;

  const int tid  = threadIdx.x;
  const int wave = tid >> 5, lane = tid & 31;
  const int lr   = lane & 15, half = lane >> 4;
  const int n0   = wave * 16;
  const int bbase = blockIdx.x * 16;

  const float* Wh = W_fc + kH * kH;                // W_fc[256:512, :]
  v16bf Breg[8];
  #pragma unroll
  for (int c = 0; c < 8; ++c)
    Breg[c] = load_b_chunk_f32(Wh, kH, c * 32 + half * 16, n0 + lr);

  // h0 = 0 (4096 bf16 = 512 threads * one 16B store).
  *(uint4*)(hbuf0 + tid * 8) = uint4{0u, 0u, 0u, 0u};
  __syncthreads();

  // Software-pipelined C operand: cin holds this step's xproj tile.
  v8f cin;
  #pragma unroll
  for (int r = 0; r < 8; ++r)
    cin[r] = xproj[((long)(bbase + r + half * 8) * kT) * kH + n0 + lr];

  v8f acc;
  #pragma unroll 1
  for (int t = 0; t < kT; ++t) {
    const __bf16* rd = (t & 1) ? hbuf1 : hbuf0;
    __bf16*       wr = (t & 1) ? hbuf0 : hbuf1;

    // Two independent accumulator chains (halves WMMA dependency depth).
    v8f acc0 = cin;
    v8f acc1 = {0.f, 0.f, 0.f, 0.f, 0.f, 0.f, 0.f, 0.f};

    // Prefetch next step's xproj tile while this step's WMMAs run.
    const int tn = (t + 1 < kT) ? (t + 1) : (kT - 1);
    #pragma unroll
    for (int r = 0; r < 8; ++r)
      cin[r] = xproj[((long)(bbase + r + half * 8) * kT + tn) * (long)kH + n0 + lr];

    // acc += h @ Wh  (K=256 as 8 chunks, 2 interleaved chains).
    const __bf16* arow = rd + lr * kH;
    #pragma unroll
    for (int c = 0; c < 8; c += 2) {
      acc0 = wmma_bf16(load_a_chunk(arow, c * 32, half),       Breg[c],     acc0);
      acc1 = wmma_bf16(load_a_chunk(arow, (c + 1) * 32, half), Breg[c + 1], acc1);
    }
    // nonlinearity (TRANS ops co-execute with WMMA pipeline)
    #pragma unroll
    for (int r = 0; r < 8; ++r) acc[r] = fast_tanh(acc0[r] + acc1[r]);

    // emit hs (f32, global) and next-step h (bf16, LDS).
    #pragma unroll
    for (int r = 0; r < 8; ++r) {
      int  m = r + half * 8;
      long b = bbase + m;
      hs[(b * kT + t) * (long)kH + n0 + lr] = acc[r];
      wr[m * kH + n0 + lr] = (__bf16)acc[r];
    }
    __syncthreads();   // next step reads wr; also protects rd for reuse
  }

  // acc still holds the t = T-1 tile in full f32.
  #pragma unroll
  for (int r = 0; r < 8; ++r)
    hlast[(bbase + r + half * 8) * kH + n0 + lr] = acc[r];
}

// ---------------------------------------------------------------------------
// Phase 3: out = h_last @ W_out + b_out.  Grid: 4 WGs (one 16-row M-tile each),
// 16 waves = 16 N-tiles.
// ---------------------------------------------------------------------------
__global__ void __launch_bounds__(512)
rnn_phase3_out(const float* __restrict__ hlast, const float* __restrict__ W_out,
               const float* __restrict__ b_out, float* __restrict__ out) {
  const int tid  = threadIdx.x;
  const int wave = tid >> 5, lane = tid & 31;
  const int lr   = lane & 15, half = lane >> 4;
  const int n0   = wave * 16;
  const int mbase = blockIdx.x * 16;

  const float bias = b_out[n0 + lr];
  v8f acc = {bias, bias, bias, bias, bias, bias, bias, bias};
  const float* arow = hlast + (mbase + lr) * kH;   // this lane's A row (f32)

  #pragma unroll
  for (int c = 0; c < 8; ++c) {
    v16bf a;
    #pragma unroll
    for (int i = 0; i < 8; ++i) {
      a[i]     = (__bf16)arow[c * 32 + half * 16 + i];
      a[i + 8] = (__bf16)arow[c * 32 + half * 16 + 8 + i];
    }
    v16bf b = load_b_chunk_f32(W_out, kOUT, c * 32 + half * 16, n0 + lr);
    acc = wmma_bf16(a, b, acc);
  }
  #pragma unroll
  for (int r = 0; r < 8; ++r)
    out[(mbase + r + half * 8) * kOUT + n0 + lr] = acc[r];
}

// ---------------------------------------------------------------------------
extern "C" void kernel_launch(void* const* d_in, const int* in_sizes, int n_in,
                              void* d_out, int out_size, void* d_ws, size_t ws_size,
                              hipStream_t stream) {
  const int*   x     = (const int*)d_in[0];
  const float* embed = (const float*)d_in[1];
  const float* W_fc  = (const float*)d_in[2];
  const float* b_fc  = (const float*)d_in[3];
  const float* W_out = (const float*)d_in[4];
  const float* b_out = (const float*)d_in[5];

  float* out = (float*)d_out;                  // (64, 256)
  float* hs  = out + (size_t)kB * kOUT;        // (64, 2048, 256)

  float* xproj = (float*)d_ws;                             // B*T*H f32 (134 MB)
  float* hlast = xproj + (size_t)kB * kT * kH;             // 64 x 256 f32

  const size_t smem1 = (size_t)128 * kH * 2 + 128 * sizeof(int);  // ~66 KB
  const size_t smem2 = (size_t)2 * 16 * kH * 2;                   // 16 KB

  rnn_phase1_xproj<<<dim3((kB * kT) / 128), dim3(512), smem1, stream>>>(
      x, embed, W_fc, b_fc, xproj);
  rnn_phase2_scan<<<dim3(kB / 16), dim3(512), smem2, stream>>>(
      W_fc, xproj, hs, hlast);
  rnn_phase3_out<<<dim3(kB / 16), dim3(512), 0, stream>>>(
      hlast, W_out, b_out, out);
}
